// Encoder_48017734369665
// MI455X (gfx1250) — matrix operand order repaired and checked by reading, not compiled, not run on hardware
//
#include <hip/hip_runtime.h>
#include <cfloat>

typedef __attribute__((ext_vector_type(16))) __bf16 bf16x16;
typedef __attribute__((ext_vector_type(8)))  __bf16 bf16x8;
typedef __attribute__((ext_vector_type(8)))  float  f32x8;

#define N_PTS   131072
#define SEGS    16
#define NT      64           // points per workgroup (4 n-tiles of 16)
#define NTT     4            // n-tiles per workgroup
#define XCS     520          // 512 + 8 pad (bf16 elems) -> 1040B row, 16B aligned, 4-bank lane step
#define HS      136          // 128 + 8 pad
#define NEG_INF (-3.402823466e38f)

// ---- workspace layout (bytes) ----
#define OFF_W2BF   0u
#define OFF_W3BF   (OFF_W2BF + 256u*128u*2u)
#define OFF_W4BF   (OFF_W3BF + 512u*512u*2u)
#define OFF_S1     (OFF_W4BF + 1024u*512u*2u)
#define OFF_T1     (OFF_S1 + 128u*4u)
#define OFF_S2     (OFF_T1 + 128u*4u)
#define OFF_T2     (OFF_S2 + 256u*4u)
#define OFF_S3     (OFF_T2 + 256u*4u)
#define OFF_T3     (OFF_S3 + 512u*4u)
#define OFF_S4     (OFF_T3 + 512u*4u)
#define OFF_T4     (OFF_S4 + 1024u*4u)
#define OFF_GBUF   (OFF_T4 + 1024u*4u)
#define OFF_FBUF   (OFF_GBUF + SEGS*256u*4u)      // fbuf: N_PTS*256 bf16 = 64MB

__device__ __forceinline__ f32x8 wmma_bf16(bf16x16 a, bf16x16 b, f32x8 c) {
  return __builtin_amdgcn_wmma_f32_16x16x32_bf16(false, a, false, b, (short)0, c, false, false);
}

__device__ __forceinline__ bf16x16 mk16(const __bf16* lo, const __bf16* hi) {
  bf16x8 a = *(const bf16x8*)lo;
  bf16x8 b = *(const bf16x8*)hi;
  bf16x16 r;
#pragma unroll
  for (int i = 0; i < 8; ++i) { r[i] = a[i]; r[i + 8] = b[i]; }
  return r;
}

__device__ __forceinline__ int seg_of(int n, const int (&cum)[16]) {
  int sg = 0;
#pragma unroll
  for (int s = 0; s < 16; ++s) sg += (n >= cum[s]) ? 1 : 0;
  return sg;
}

// ---------------- Kernel 0: weight convert + BN fold + init ----------------
__global__ void enc_prep(
    const float* __restrict__ W2, const float* __restrict__ W3, const float* __restrict__ W4,
    const float* __restrict__ b1, const float* __restrict__ g1, const float* __restrict__ be1,
    const float* __restrict__ m1, const float* __restrict__ v1,
    const float* __restrict__ b2, const float* __restrict__ g2, const float* __restrict__ be2,
    const float* __restrict__ m2, const float* __restrict__ v2,
    const float* __restrict__ b3, const float* __restrict__ g3, const float* __restrict__ be3,
    const float* __restrict__ m3, const float* __restrict__ v3,
    const float* __restrict__ b4, const float* __restrict__ g4, const float* __restrict__ be4,
    const float* __restrict__ m4, const float* __restrict__ v4,
    __bf16* __restrict__ W2bf, __bf16* __restrict__ W3bf, __bf16* __restrict__ W4bf,
    float* __restrict__ s1, float* __restrict__ t1, float* __restrict__ s2, float* __restrict__ t2,
    float* __restrict__ s3, float* __restrict__ t3, float* __restrict__ s4, float* __restrict__ t4,
    float* __restrict__ gbuf, float* __restrict__ outv)
{
  const int idx = blockIdx.x * 256 + threadIdx.x;         // grid = 2048*256 = 524288
  if (idx < 256 * 128) W2bf[idx] = (__bf16)W2[idx];
  if (idx < 512 * 512) W3bf[idx] = (__bf16)W3[idx];
  W4bf[idx] = (__bf16)W4[idx];                            // exactly 1024*512
  if (idx < 128)  { float s = g1[idx] * rsqrtf(v1[idx] + 1e-5f); s1[idx] = s; t1[idx] = (b1[idx] - m1[idx]) * s + be1[idx]; }
  if (idx < 256)  { float s = g2[idx] * rsqrtf(v2[idx] + 1e-5f); s2[idx] = s; t2[idx] = (b2[idx] - m2[idx]) * s + be2[idx]; }
  if (idx < 512)  { float s = g3[idx] * rsqrtf(v3[idx] + 1e-5f); s3[idx] = s; t3[idx] = (b3[idx] - m3[idx]) * s + be3[idx]; }
  if (idx < 1024) { float s = g4[idx] * rsqrtf(v4[idx] + 1e-5f); s4[idx] = s; t4[idx] = (b4[idx] - m4[idx]) * s + be4[idx]; }
  if (idx < SEGS * 256)  gbuf[idx] = NEG_INF;
  if (idx < SEGS * 1024) outv[idx] = NEG_INF;
}

// ---------------- Kernel 1: layer1 (VALU) + layer2 (WMMA) + g segment-max ----------------
__global__ void __launch_bounds__(256) enc_l12(
    const float* __restrict__ x, const int* __restrict__ npts,
    const float* __restrict__ W1,
    const float* __restrict__ s1, const float* __restrict__ t1,
    const __bf16* __restrict__ W2bf,
    const float* __restrict__ s2, const float* __restrict__ t2,
    __bf16* __restrict__ fbuf, float* __restrict__ gbuf)
{
  extern __shared__ __align__(16) char smem[];
  __bf16* h = (__bf16*)smem;                               // NT x HS bf16
  const int tid  = threadIdx.x;
  const int base = blockIdx.x * NT;

  int cum[16];
  { int c = 0;
#pragma unroll
    for (int s = 0; s < 16; ++s) { c += npts[s]; cum[s] = c; } }

  // layer1: 128x3 GEMM + BN + relu, point-major into LDS
#pragma unroll
  for (int it = 0; it < (NT * 128) / 256; ++it) {
    int idx = it * 256 + tid;
    int p = idx >> 7, ch = idx & 127;
    int n = base + p;
    float z = W1[ch * 3 + 0] * x[n] + W1[ch * 3 + 1] * x[N_PTS + n] + W1[ch * 3 + 2] * x[2 * N_PTS + n];
    h[p * HS + ch] = (__bf16)fmaxf(z * s1[ch] + t1[ch], 0.0f);
  }
  __syncthreads();

  const int wave = tid >> 5, lane = tid & 31;
  const int col = lane & 15, half = lane >> 4;
  const int koff = half * 8, mbase = half * 8;
  const int segW0 = seg_of(base, cum), segW1 = seg_of(base + NT - 1, cum);
  const bool wgU = (segW0 == segW1);

  for (int mt = wave; mt < 16; mt += 8) {                  // 256 out channels / 16
    f32x8 acc[NTT] = {};
    const __bf16* wr = W2bf + (size_t)(mt * 16 + col) * 128;
#pragma unroll 2
    for (int kk = 0; kk < 128; kk += 32) {
      bf16x16 a = mk16(wr + kk + koff, wr + kk + koff + 16);
#pragma unroll
      for (int nt = 0; nt < NTT; ++nt) {
        const __bf16* xr = h + (nt * 16 + col) * HS + kk + half * 8;
        acc[nt] = wmma_bf16(a, mk16(xr, xr + 16), acc[nt]);
      }
    }
    const int m0 = mt * 16 + mbase;
    float mx[8];
#pragma unroll
    for (int i = 0; i < 8; ++i) mx[i] = NEG_INF;
#pragma unroll
    for (int nt = 0; nt < NTT; ++nt) {
      float vals[8]; bf16x8 fv;
#pragma unroll
      for (int i = 0; i < 8; ++i) { float y = acc[nt][i] * s2[m0 + i] + t2[m0 + i]; vals[i] = y; fv[i] = (__bf16)y; }
      const int n = base + nt * 16 + col;
      *(bf16x8*)&fbuf[(size_t)n * 256 + m0] = fv;          // f for kernel 2
      const int sgF = seg_of(base + nt * 16, cum), sgL = seg_of(base + nt * 16 + 15, cum);
      if (sgF == sgL) {
#pragma unroll
        for (int s = 1; s < 16; s <<= 1)
#pragma unroll
          for (int i = 0; i < 8; ++i) vals[i] = fmaxf(vals[i], __shfl_xor(vals[i], s, 32));
        if (wgU) {
#pragma unroll
          for (int i = 0; i < 8; ++i) mx[i] = fmaxf(mx[i], vals[i]);
        } else if (col == 0) {
#pragma unroll
          for (int i = 0; i < 8; ++i) atomicMax(&gbuf[sgF * 256 + m0 + i], vals[i]);
        }
      } else {                                             // rare segment-boundary tile
        const int sp = seg_of(n, cum);
#pragma unroll
        for (int i = 0; i < 8; ++i) atomicMax(&gbuf[sp * 256 + m0 + i], vals[i]);
      }
    }
    if (wgU && col == 0) {
#pragma unroll
      for (int i = 0; i < 8; ++i) atomicMax(&gbuf[segW0 * 256 + m0 + i], mx[i]);
    }
  }
}

// ---------------- Kernel 2: build xc, layer3 + layer4 (WMMA), v segment-max ----------------
__global__ void __launch_bounds__(256) enc_l34(
    const int* __restrict__ npts,
    const __bf16* __restrict__ fbuf, const float* __restrict__ gbuf,
    const __bf16* __restrict__ W3bf, const float* __restrict__ s3, const float* __restrict__ t3,
    const __bf16* __restrict__ W4bf, const float* __restrict__ s4, const float* __restrict__ t4,
    float* __restrict__ outv)
{
  extern __shared__ __align__(16) char smem[];
  __bf16* xc = (__bf16*)smem;                              // NT x XCS
  __bf16* h3 = xc + NT * XCS;                              // NT x XCS
  const int tid  = threadIdx.x;
  const int base = blockIdx.x * NT;

  int cum[16];
  { int c = 0;
#pragma unroll
    for (int s = 0; s < 16; ++s) { c += npts[s]; cum[s] = c; } }

  // xc[p][256..511] = f (16B chunks)
#pragma unroll
  for (int it = 0; it < (NT * 32) / 256; ++it) {
    int idx = it * 256 + tid;
    int p = idx >> 5, ch = idx & 31;
    *(bf16x8*)&xc[p * XCS + 256 + ch * 8] = *(const bf16x8*)&fbuf[(size_t)(base + p) * 256 + ch * 8];
  }
  // xc[p][0..255] = g[seg(p)] broadcast (seg uniform per p -> scalar)
  for (int p = 0; p < NT; ++p) {
    const int sg = seg_of(base + p, cum);
    xc[p * XCS + tid] = (__bf16)gbuf[sg * 256 + tid];
  }
  __syncthreads();

  const int wave = tid >> 5, lane = tid & 31;
  const int col = lane & 15, half = lane >> 4;
  const int koff = half * 8, mbase = half * 8;

  // layer3: 512x512, relu, into LDS h3 (point-major bf16)
  for (int mt = wave; mt < 32; mt += 8) {
    f32x8 acc[NTT] = {};
    const __bf16* wr = W3bf + (size_t)(mt * 16 + col) * 512;
#pragma unroll 2
    for (int kk = 0; kk < 512; kk += 32) {
      bf16x16 a = mk16(wr + kk + koff, wr + kk + koff + 16);
#pragma unroll
      for (int nt = 0; nt < NTT; ++nt) {
        const __bf16* xr = xc + (nt * 16 + col) * XCS + kk + half * 8;
        acc[nt] = wmma_bf16(a, mk16(xr, xr + 16), acc[nt]);
      }
    }
    const int m0 = mt * 16 + mbase;
#pragma unroll
    for (int nt = 0; nt < NTT; ++nt) {
      bf16x8 hv;
#pragma unroll
      for (int i = 0; i < 8; ++i) hv[i] = (__bf16)fmaxf(acc[nt][i] * s3[m0 + i] + t3[m0 + i], 0.0f);
      *(bf16x8*)&h3[(nt * 16 + col) * XCS + m0] = hv;
    }
  }
  __syncthreads();

  // layer4: 1024x512, BN, ragged segment max -> outv
  const int segW0 = seg_of(base, cum), segW1 = seg_of(base + NT - 1, cum);
  const bool wgU = (segW0 == segW1);
  for (int mt = wave; mt < 64; mt += 8) {
    f32x8 acc[NTT] = {};
    const __bf16* wr = W4bf + (size_t)(mt * 16 + col) * 512;
#pragma unroll 2
    for (int kk = 0; kk < 512; kk += 32) {
      bf16x16 a = mk16(wr + kk + koff, wr + kk + koff + 16);
#pragma unroll
      for (int nt = 0; nt < NTT; ++nt) {
        const __bf16* xr = h3 + (nt * 16 + col) * XCS + kk + half * 8;
        acc[nt] = wmma_bf16(a, mk16(xr, xr + 16), acc[nt]);
      }
    }
    const int m0 = mt * 16 + mbase;
    float mx[8];
#pragma unroll
    for (int i = 0; i < 8; ++i) mx[i] = NEG_INF;
#pragma unroll
    for (int nt = 0; nt < NTT; ++nt) {
      float vals[8];
#pragma unroll
      for (int i = 0; i < 8; ++i) vals[i] = acc[nt][i] * s4[m0 + i] + t4[m0 + i];
      const int sgF = seg_of(base + nt * 16, cum), sgL = seg_of(base + nt * 16 + 15, cum);
      if (sgF == sgL) {
#pragma unroll
        for (int s = 1; s < 16; s <<= 1)
#pragma unroll
          for (int i = 0; i < 8; ++i) vals[i] = fmaxf(vals[i], __shfl_xor(vals[i], s, 32));
        if (wgU) {
#pragma unroll
          for (int i = 0; i < 8; ++i) mx[i] = fmaxf(mx[i], vals[i]);
        } else if (col == 0) {
#pragma unroll
          for (int i = 0; i < 8; ++i) atomicMax(&outv[sgF * 1024 + m0 + i], vals[i]);
        }
      } else {
        const int sp = seg_of(base + nt * 16 + col, cum);
#pragma unroll
        for (int i = 0; i < 8; ++i) atomicMax(&outv[sp * 1024 + m0 + i], vals[i]);
      }
    }
    if (wgU && col == 0) {
#pragma unroll
      for (int i = 0; i < 8; ++i) atomicMax(&outv[segW0 * 1024 + m0 + i], mx[i]);
    }
  }
}

extern "C" void kernel_launch(void* const* d_in, const int* in_sizes, int n_in,
                              void* d_out, int out_size, void* d_ws, size_t ws_size,
                              hipStream_t stream) {
  (void)in_sizes; (void)n_in; (void)out_size; (void)ws_size;
  const float* x   = (const float*)d_in[0];
  const int*   np  = (const int*)  d_in[1];
  const float* W1  = (const float*)d_in[2];
  const float* b1  = (const float*)d_in[3];
  const float* g1  = (const float*)d_in[4];
  const float* be1 = (const float*)d_in[5];
  const float* m1  = (const float*)d_in[6];
  const float* v1  = (const float*)d_in[7];
  const float* W2  = (const float*)d_in[8];
  const float* b2  = (const float*)d_in[9];
  const float* g2  = (const float*)d_in[10];
  const float* be2 = (const float*)d_in[11];
  const float* m2  = (const float*)d_in[12];
  const float* v2  = (const float*)d_in[13];
  const float* W3  = (const float*)d_in[14];
  const float* b3  = (const float*)d_in[15];
  const float* g3  = (const float*)d_in[16];
  const float* be3 = (const float*)d_in[17];
  const float* m3  = (const float*)d_in[18];
  const float* v3  = (const float*)d_in[19];
  const float* W4  = (const float*)d_in[20];
  const float* b4  = (const float*)d_in[21];
  const float* g4  = (const float*)d_in[22];
  const float* be4 = (const float*)d_in[23];
  const float* m4  = (const float*)d_in[24];
  const float* v4  = (const float*)d_in[25];

  char* ws = (char*)d_ws;
  __bf16* W2bf = (__bf16*)(ws + OFF_W2BF);
  __bf16* W3bf = (__bf16*)(ws + OFF_W3BF);
  __bf16* W4bf = (__bf16*)(ws + OFF_W4BF);
  float*  s1 = (float*)(ws + OFF_S1); float* t1 = (float*)(ws + OFF_T1);
  float*  s2 = (float*)(ws + OFF_S2); float* t2 = (float*)(ws + OFF_T2);
  float*  s3 = (float*)(ws + OFF_S3); float* t3 = (float*)(ws + OFF_T3);
  float*  s4 = (float*)(ws + OFF_S4); float* t4 = (float*)(ws + OFF_T4);
  float*  gbuf = (float*)(ws + OFF_GBUF);
  __bf16* fbuf = (__bf16*)(ws + OFF_FBUF);
  float*  outv = (float*)d_out;

  enc_prep<<<2048, 256, 0, stream>>>(W2, W3, W4,
      b1, g1, be1, m1, v1, b2, g2, be2, m2, v2,
      b3, g3, be3, m3, v3, b4, g4, be4, m4, v4,
      W2bf, W3bf, W4bf, s1, t1, s2, t2, s3, t3, s4, t4, gbuf, outv);

  enc_l12<<<N_PTS / NT, 256, NT * HS * 2, stream>>>(
      x, np, W1, s1, t1, W2bf, s2, t2, fbuf, gbuf);

  enc_l34<<<N_PTS / NT, 256, 2 * NT * XCS * 2, stream>>>(
      np, fbuf, gbuf, W3bf, s3, t3, W4bf, s4, t4, outv);
}
